// ActorCritic_74835510165516
// MI455X (gfx1250) — compile-verified, matrix operand order checked by reading, not run on hardware
//
#include <hip/hip_runtime.h>
#include <hip/hip_bf16.h>
#include <cstdint>

// Model dims (match reference)
#define BB 256
#define NN 256
#define EE 128
#define HH 128
#define FFD 512
#define NHD 8
#define DHD 16
#define LL 3

typedef __attribute__((ext_vector_type(16))) _Float16 v16h;
typedef __attribute__((ext_vector_type(8)))  float    v8f;

// ---------------- WMMA fragment helpers (CDNA5 16x16x32 f16 layouts) ----------------
// A (16x32, MxK): lane&15 = M; k = 8*(lane>>4) + 16*(i>>3) + (i&7)
__device__ __forceinline__ int frag_k(int lane, int i) {
  return ((lane >> 4) << 3) + ((i >> 3) << 4) + (i & 7);
}

__device__ __forceinline__ v16h load_a_f16(const _Float16* base, int ld, int lane) {
  v16h a; int row = lane & 15;
#pragma unroll
  for (int i = 0; i < 16; ++i) a[i] = base[row * ld + frag_k(lane, i)];
  return a;
}

// A fragment sourced from f32 LDS (converted on the fly)
__device__ __forceinline__ v16h load_a_f32(const float* base, int ld, int lane) {
  v16h a; int row = lane & 15;
#pragma unroll
  for (int i = 0; i < 16; ++i) a[i] = (_Float16)base[row * ld + frag_k(lane, i)];
  return a;
}

// A fragment with only 16 valid K (zero-pad K=16..31)
__device__ __forceinline__ v16h load_a_pad16(const _Float16* base, int ld, int lane) {
  v16h a; int row = lane & 15;
#pragma unroll
  for (int i = 0; i < 16; ++i) {
    int k = frag_k(lane, i);
    a[i] = (k < 16) ? base[row * ld + k] : (_Float16)0.f;
  }
  return a;
}

// B (32x16, KxN) stored as [n][k] rows (torch Linear weight layout)
__device__ __forceinline__ v16h load_b_nk(const _Float16* base, int ld, int lane) {
  v16h b; int col = lane & 15;
#pragma unroll
  for (int i = 0; i < 16; ++i) b[i] = base[col * ld + frag_k(lane, i)];
  return b;
}

__device__ __forceinline__ v16h load_b_nk_pad16(const _Float16* base, int ld, int lane) {
  v16h b; int col = lane & 15;
#pragma unroll
  for (int i = 0; i < 16; ++i) {
    int k = frag_k(lane, i);
    b[i] = (k < 16) ? base[col * ld + k] : (_Float16)0.f;
  }
  return b;
}

// B stored as [k][n]
__device__ __forceinline__ v16h load_b_kn(const _Float16* base, int ld, int lane) {
  v16h b; int col = lane & 15;
#pragma unroll
  for (int i = 0; i < 16; ++i) b[i] = base[frag_k(lane, i) * ld + col];
  return b;
}

// ---------------- Tiled WMMA GEMM: C[M,N] = act(A[M,K] @ W[N,K]^T + bias[N]) ----------------
// M%64==0, N%64==0, K%32==0. Block 256 threads (8 waves). 64x64 C tile,
// each wave owns a 16x32 strip (2 accumulators).
template <int ACT> // 0 none, 1 relu, 2 tanh
__global__ __launch_bounds__(256) void gemm_bias(const float* __restrict__ A,
                                                 const float* __restrict__ W,
                                                 const float* __restrict__ bias,
                                                 float* __restrict__ C,
                                                 int M, int N, int K) {
  __shared__ _Float16 As[64][32];
  __shared__ _Float16 Bs[64][32];
  const int tid = threadIdx.x;
  const int wave = tid >> 5, lane = tid & 31;
  const int m0 = blockIdx.x * 64, n0 = blockIdx.y * 64;
  const int wr = wave >> 1;          // row strip 0..3
  const int wc = (wave & 1) << 1;    // col subtile base (16-col units): 0 or 2
  v8f acc0 = {}, acc1 = {};
  const int lr = tid >> 2, lc = (tid & 3) * 8;
  for (int k0 = 0; k0 < K; k0 += 32) {
    const float* ap = A + (size_t)(m0 + lr) * K + k0 + lc;
    const float* wp = W + (size_t)(n0 + lr) * K + k0 + lc;
#pragma unroll
    for (int j = 0; j < 8; ++j) As[lr][lc + j] = (_Float16)ap[j];
#pragma unroll
    for (int j = 0; j < 8; ++j) Bs[lr][lc + j] = (_Float16)wp[j];
    __syncthreads();
    v16h af = load_a_f16(&As[wr * 16][0], 32, lane);
    v16h b0 = load_b_nk(&Bs[wc * 16][0], 32, lane);
    v16h b1 = load_b_nk(&Bs[(wc + 1) * 16][0], 32, lane);
    acc0 = __builtin_amdgcn_wmma_f32_16x16x32_f16(false, af, false, b0, (short)0, acc0, false, false);
    acc1 = __builtin_amdgcn_wmma_f32_16x16x32_f16(false, af, false, b1, (short)0, acc1, false, false);
    __syncthreads();
  }
  const int mloc = (lane >> 4) << 3, ncol = lane & 15;
#pragma unroll
  for (int r = 0; r < 8; ++r) {
    int m = m0 + wr * 16 + mloc + r;
    int n = n0 + wc * 16 + ncol;
    float v0 = acc0[r] + bias[n];
    float v1 = acc1[r] + bias[n + 16];
    if (ACT == 1) { v0 = fmaxf(v0, 0.f); v1 = fmaxf(v1, 0.f); }
    if (ACT == 2) { v0 = tanhf(v0); v1 = tanhf(v1); }
    C[(size_t)m * N + n] = v0;
    C[(size_t)m * N + n + 16] = v1;
  }
}

// ---------------- Input embedding: x0 = coords @ Wemb^T + bemb (K=2) ----------------
__global__ __launch_bounds__(256) void embed_kernel(const float* __restrict__ coords,
                                                    const float* __restrict__ Wemb,
                                                    const float* __restrict__ bemb,
                                                    float* __restrict__ x0,
                                                    float* __restrict__ h) {
  int i = blockIdx.x * 256 + threadIdx.x; // over B*N*E
  int bn = i >> 7, e = i & 127;
  float c0 = coords[bn * 2], c1 = coords[bn * 2 + 1];
  float v = c0 * Wemb[e * 2] + c1 * Wemb[e * 2 + 1] + bemb[e];
  x0[i] = v;
  h[i] = v;
}

// ---------------- Attention: per (b, head, 32-row q-tile) ----------------
__global__ __launch_bounds__(256) void attention_kernel(const float* __restrict__ qkv,
                                                        float* __restrict__ o) {
  __shared__ _Float16 Qh[32][16];
  __shared__ _Float16 Kh[256][16];
  __shared__ _Float16 Vh[256][16];
  __shared__ float    S[32][256];
  const int bid = blockIdx.x;
  const int qt = bid & 7;
  const int hd = (bid >> 3) & 7;
  const int b  = bid >> 6;
  const int tid = threadIdx.x;
  const int wave = tid >> 5, lane = tid & 31;
  const float* base = qkv + (size_t)b * NN * (3 * EE);
  // K, V: 256 rows x 16, one row per thread
  {
    const float* kp = base + (size_t)tid * 384 + EE + hd * 16;
    const float* vp = base + (size_t)tid * 384 + 2 * EE + hd * 16;
#pragma unroll
    for (int d = 0; d < 16; ++d) { Kh[tid][d] = (_Float16)kp[d]; Vh[tid][d] = (_Float16)vp[d]; }
  }
  // Q tile: 32 rows x 16
  {
    int r = tid >> 3, d = (tid & 7) * 2;
    const float* qp = base + (size_t)(qt * 32 + r) * 384 + hd * 16 + d;
    Qh[r][d] = (_Float16)qp[0];
    Qh[r][d + 1] = (_Float16)qp[1];
  }
  __syncthreads();
  // scores: 2 row strips x 16 col subtiles; wave -> strip (wave>>2), 4 col subtiles each
  {
    const int wr = wave >> 2;
    const int cb = (wave & 3) * 64;
    v16h af = load_a_pad16(&Qh[wr * 16][0], 16, lane);
    const int mloc = (lane >> 4) << 3, ncol = lane & 15;
#pragma unroll
    for (int s = 0; s < 4; ++s) {
      int nc = cb + s * 16;
      v16h bf = load_b_nk_pad16(&Kh[nc][0], 16, lane);
      v8f acc = {};
      acc = __builtin_amdgcn_wmma_f32_16x16x32_f16(false, af, false, bf, (short)0, acc, false, false);
#pragma unroll
      for (int r = 0; r < 8; ++r) S[wr * 16 + mloc + r][nc + ncol] = acc[r] * 0.25f;
    }
  }
  __syncthreads();
  // softmax: 8 threads per q-row, width-8 shuffle reductions
  {
    const int row = tid >> 3, seg = tid & 7;
    float mx = -INFINITY;
#pragma unroll
    for (int j = 0; j < 32; ++j) mx = fmaxf(mx, S[row][seg * 32 + j]);
#pragma unroll
    for (int off = 4; off; off >>= 1) mx = fmaxf(mx, __shfl_xor(mx, off, 8));
    float sum = 0.f;
#pragma unroll
    for (int j = 0; j < 32; ++j) {
      float e = __expf(S[row][seg * 32 + j] - mx);
      S[row][seg * 32 + j] = e; sum += e;
    }
#pragma unroll
    for (int off = 4; off; off >>= 1) sum += __shfl_xor(sum, off, 8);
    float inv = 1.f / sum;
#pragma unroll
    for (int j = 0; j < 32; ++j) S[row][seg * 32 + j] *= inv;
  }
  __syncthreads();
  // O = P @ V : 2 strips of 16x16, waves 0..1 (wave-uniform branch)
  if (wave < 2) {
    v8f acc = {};
#pragma unroll
    for (int kc = 0; kc < 8; ++kc) {
      v16h af = load_a_f32(&S[wave * 16][kc * 32], 256, lane);
      v16h bf = load_b_kn(&Vh[kc * 32][0], 16, lane);
      acc = __builtin_amdgcn_wmma_f32_16x16x32_f16(false, af, false, bf, (short)0, acc, false, false);
    }
    const int mloc = (lane >> 4) << 3, ncol = lane & 15;
#pragma unroll
    for (int r = 0; r < 8; ++r) {
      int qrow = qt * 32 + wave * 16 + mloc + r;
      o[(size_t)b * NN * EE + (size_t)qrow * EE + hd * 16 + ncol] = acc[r];
    }
  }
}

// ---------------- Fused residual + LayerNorm ----------------
__global__ __launch_bounds__(256) void add_ln_kernel(const float* __restrict__ X,
                                                     const float* __restrict__ Y,
                                                     const float* __restrict__ g,
                                                     const float* __restrict__ bt,
                                                     float* __restrict__ out) {
  const int wave = threadIdx.x >> 5, lane = threadIdx.x & 31;
  const size_t row = (size_t)blockIdx.x * 8 + wave;
  const float* x = X + row * EE;
  const float* y = Y + row * EE;
  float v[4]; float s = 0.f;
#pragma unroll
  for (int j = 0; j < 4; ++j) { v[j] = x[lane * 4 + j] + y[lane * 4 + j]; s += v[j]; }
#pragma unroll
  for (int off = 16; off; off >>= 1) s += __shfl_xor(s, off, 32);
  float mu = s * (1.f / 128.f);
  float var = 0.f;
#pragma unroll
  for (int j = 0; j < 4; ++j) { float d = v[j] - mu; var += d * d; }
#pragma unroll
  for (int off = 16; off; off >>= 1) var += __shfl_xor(var, off, 32);
  float inv = rsqrtf(var * (1.f / 128.f) + 1e-5f);
#pragma unroll
  for (int j = 0; j < 4; ++j)
    out[row * EE + lane * 4 + j] = (v[j] - mu) * inv * g[lane * 4 + j] + bt[lane * 4 + j];
}

// ---------------- Elementwise add ----------------
__global__ __launch_bounds__(256) void add_kernel(const float* __restrict__ a,
                                                  const float* __restrict__ b,
                                                  float* __restrict__ c) {
  size_t i = (size_t)blockIdx.x * 256 + threadIdx.x;
  c[i] = a[i] + b[i];
}

// ---------------- pooled = mean over N ----------------
__global__ __launch_bounds__(256) void mean_kernel(const float* __restrict__ emb,
                                                   float* __restrict__ pooled) {
  int i = blockIdx.x * 256 + threadIdx.x; // over B*E
  int b = i >> 7, e = i & 127;
  float s = 0.f;
  const float* p = emb + (size_t)b * NN * EE + e;
  for (int n = 0; n < NN; ++n) s += p[(size_t)n * EE];
  pooled[i] = s * (1.f / (float)NN);
}

// ---------------- value head final ----------------
__global__ __launch_bounds__(256) void value_kernel(const float* __restrict__ hid1,
                                                    const float* __restrict__ v2w,
                                                    const float* __restrict__ v2b,
                                                    float* __restrict__ values) {
  int b = threadIdx.x;
  float s = v2b[0];
  const float* hp = hid1 + b * 64;
  for (int j = 0; j < 64; ++j) s += hp[j] * v2w[j];
  values[b] = s;
}

// ---------------- small helpers for decode precompute ----------------
__global__ __launch_bounds__(256) void transpose128_kernel(const float* __restrict__ in,
                                                           float* __restrict__ out) {
  int i = blockIdx.x * 256 + threadIdx.x; // 128*128
  int r = i >> 7, c = i & 127;
  out[c * 128 + r] = in[i];
}

__global__ __launch_bounds__(128) void fill_zero_kernel(float* __restrict__ p) {
  p[threadIdx.x] = 0.f;
}

// ---------------- Greedy GRU pointer decoder: one workgroup per batch ----------------
// Uses precomputed PRE  = node_emb @ h2q_w        (B,N,H)  -> scores[n] = scale * hidden.PRE[n]
//                  GIPRE = node_emb @ wih^T + bih (B,N,3H) -> gi = GIPRE[idx]
__global__ __launch_bounds__(128) void decode_kernel(const float* __restrict__ pre,   // (B,N,H)
                                                     const float* __restrict__ gipre, // (B,N,3H)
                                                     const float* __restrict__ whh,   // (3H,H)
                                                     const float* __restrict__ bhh,   // (3H)
                                                     const float* __restrict__ hid0,  // (B,H)
                                                     float* __restrict__ tours,
                                                     float* __restrict__ logp,
                                                     float* __restrict__ ent) {
  __shared__ float hidden[HH];
  __shared__ float sc[NN];
  __shared__ int   visited[NN];
  __shared__ float redv[4];
  __shared__ int   redi[4];
  __shared__ float reds[4];
  __shared__ float rede[4];
  const int b = blockIdx.x, t = threadIdx.x;
  const int wave = t >> 5, lane = t & 31;
  const float* prb = pre + (size_t)b * NN * HH;
  const float* gib = gipre + (size_t)b * NN * (3 * HH);
  hidden[t] = hid0[b * HH + t];
  visited[t] = 0; visited[t + 128] = 0;
  float acc_lp = 0.f, acc_ent = 0.f;
  __syncthreads();
  const float scale = 0.08838834764831845f; // 1/sqrt(E)
  for (int step = 0; step < NN; ++step) {
    // ---- Phase A: scores for n=t and n=t+128 (4-way partial accumulation) + argmax ----
    const float* p0 = prb + (size_t)t * HH;
    const float* p1 = p0 + (size_t)128 * HH;
    float a0 = 0.f, a1 = 0.f, a2 = 0.f, a3 = 0.f;
    float c0 = 0.f, c1 = 0.f, c2 = 0.f, c3 = 0.f;
    for (int j = 0; j < HH; j += 4) {
      float h0 = hidden[j], h1 = hidden[j + 1], h2 = hidden[j + 2], h3 = hidden[j + 3];
      a0 += h0 * p0[j];     a1 += h1 * p0[j + 1];
      a2 += h2 * p0[j + 2]; a3 += h3 * p0[j + 3];
      c0 += h0 * p1[j];     c1 += h1 * p1[j + 1];
      c2 += h2 * p1[j + 2]; c3 += h3 * p1[j + 3];
    }
    float s0 = visited[t]       ? -INFINITY : ((a0 + a1) + (a2 + a3)) * scale;
    float s1 = visited[t + 128] ? -INFINITY : ((c0 + c1) + (c2 + c3)) * scale;
    sc[t] = s0; sc[t + 128] = s1;
    float mv = s0; int mi = t;
    if (s1 > mv) { mv = s1; mi = t + 128; }
#pragma unroll
    for (int off = 16; off; off >>= 1) {
      float ov = __shfl_xor(mv, off, 32);
      int   oi = __shfl_xor(mi, off, 32);
      if (ov > mv || (ov == mv && oi < mi)) { mv = ov; mi = oi; }
    }
    if (lane == 0) { redv[wave] = mv; redi[wave] = mi; }
    __syncthreads(); // B1
    // ---- Phase B: global max/argmax, exp, partial sums ----
    float mx = redv[0]; int idx = redi[0];
#pragma unroll
    for (int w = 1; w < 4; ++w) {
      float ov = redv[w]; int oi = redi[w];
      if (ov > mx || (ov == mx && oi < idx)) { mx = ov; idx = oi; }
    }
    float e0 = __expf(sc[t] - mx);
    float e1 = __expf(sc[t + 128] - mx);
    sc[t] = e0; sc[t + 128] = e1;
    float part = e0 + e1;
#pragma unroll
    for (int off = 16; off; off >>= 1) part += __shfl_xor(part, off, 32);
    if (lane == 0) reds[wave] = part;
    __syncthreads(); // B2
    // ---- Phase C: probs + entropy partial ----
    float inv = 1.f / (((reds[0] + reds[1]) + (reds[2] + reds[3])));
    float pp0 = e0 * inv, pp1 = e1 * inv;
    sc[t] = pp0; sc[t + 128] = pp1;
    float pent = pp0 * __logf(pp0 + 1e-12f) + pp1 * __logf(pp1 + 1e-12f);
#pragma unroll
    for (int off = 16; off; off >>= 1) pent += __shfl_xor(pent, off, 32);
    if (lane == 0) rede[wave] = pent;
    __syncthreads(); // B3
    // ---- Phase D: bookkeeping + GRU update (gh in registers, gi from GIPRE) ----
    acc_ent += -(((rede[0] + rede[1]) + (rede[2] + rede[3])));
    if (t == 0) {
      acc_lp += __logf(sc[idx] + 1e-12f);
      tours[(size_t)b * NN + step] = (float)idx;
      visited[idx] = 1;
    }
    const float* gir = gib + (size_t)idx * (3 * HH);
    float gh[3];
#pragma unroll
    for (int rr = 0; rr < 3; ++rr) {
      int row = t + rr * HH;
      const float* wh = whh + (size_t)row * HH;
      float d0 = 0.f, d1 = 0.f, d2 = 0.f, d3 = 0.f;
      for (int j = 0; j < HH; j += 4) {
        d0 += wh[j] * hidden[j];
        d1 += wh[j + 1] * hidden[j + 1];
        d2 += wh[j + 2] * hidden[j + 2];
        d3 += wh[j + 3] * hidden[j + 3];
      }
      gh[rr] = bhh[row] + ((d0 + d1) + (d2 + d3));
    }
    float gi_r = gir[t], gi_z = gir[HH + t], gi_n = gir[2 * HH + t];
    float r = 1.f / (1.f + __expf(-(gi_r + gh[0])));
    float z = 1.f / (1.f + __expf(-(gi_z + gh[1])));
    float nn2 = tanhf(gi_n + r * gh[2]);
    float hnew = (1.f - z) * nn2 + z * hidden[t];
    __syncthreads(); // B4: everyone done reading hidden
    hidden[t] = hnew;
    __syncthreads(); // B5: hidden/visited visible for next step
  }
  if (t == 0) { logp[b] = acc_lp; ent[b] = acc_ent; }
}

// ---------------- Host launcher ----------------
extern "C" void kernel_launch(void* const* d_in, const int* in_sizes, int n_in,
                              void* d_out, int out_size, void* d_ws, size_t ws_size,
                              hipStream_t stream) {
  (void)in_sizes; (void)n_in; (void)out_size; (void)ws_size;
  const float* coords  = (const float*)d_in[0];
  const float* Wemb    = (const float*)d_in[1];
  const float* bemb    = (const float*)d_in[2];
  const float* qkv_w   = (const float*)d_in[3];
  const float* qkv_b   = (const float*)d_in[4];
  const float* ao_w    = (const float*)d_in[5];
  const float* ao_b    = (const float*)d_in[6];
  const float* ln1_g   = (const float*)d_in[7];
  const float* ln1_b   = (const float*)d_in[8];
  const float* ff1_w   = (const float*)d_in[9];
  const float* ff1_b   = (const float*)d_in[10];
  const float* ff2_w   = (const float*)d_in[11];
  const float* ff2_b   = (const float*)d_in[12];
  const float* ln2_g   = (const float*)d_in[13];
  const float* ln2_b   = (const float*)d_in[14];
  const float* post1_w = (const float*)d_in[15];
  const float* post1_b = (const float*)d_in[16];
  const float* post2_w = (const float*)d_in[17];
  const float* post2_b = (const float*)d_in[18];
  const float* h2q_w   = (const float*)d_in[19];
  const float* p2h_w   = (const float*)d_in[20];
  const float* p2h_b   = (const float*)d_in[21];
  const float* gru_wih = (const float*)d_in[22];
  const float* gru_whh = (const float*)d_in[23];
  const float* gru_bih = (const float*)d_in[24];
  const float* gru_bhh = (const float*)d_in[25];
  const float* v1_w    = (const float*)d_in[26];
  const float* v1_b    = (const float*)d_in[27];
  const float* v2_w    = (const float*)d_in[28];
  const float* v2_b    = (const float*)d_in[29];

  float* out = (float*)d_out;
  const size_t BN  = (size_t)BB * NN;        // 65536
  const size_t BNE = BN * EE;                // 8.39M

  // Workspace layout (floats)
  float* ws     = (float*)d_ws;
  float* p_x0   = ws;
  float* p_h    = p_x0 + BNE;                // ends as node_emb
  float* p_tmp  = p_h + BNE;                 // ends as PRE (B,N,H)
  float* p_big  = p_tmp + BNE;               // BN * FF; ends as GIPRE (B,N,3H)
  float* p_pool = p_big + BN * FFD;
  float* p_hid  = p_pool + (size_t)BB * EE;
  float* p_v1   = p_hid + (size_t)BB * HH;
  float* p_h2qT = p_v1 + (size_t)BB * 64;
  float* p_zero = p_h2qT + (size_t)EE * HH;

  const int M = (int)BN;

  // x0 = coords @ Wemb^T + bemb ; h = x0
  embed_kernel<<<(int)(BNE / 256), 256, 0, stream>>>(coords, Wemb, bemb, p_x0, p_h);

  for (int l = 0; l < LL; ++l) {
    gemm_bias<0><<<dim3(M / 64, 384 / 64), 256, 0, stream>>>(
        p_h, qkv_w + (size_t)l * 384 * EE, qkv_b + (size_t)l * 384, p_big, M, 384, EE);
    attention_kernel<<<BB * NHD * (NN / 32), 256, 0, stream>>>(p_big, p_tmp);
    gemm_bias<0><<<dim3(M / 64, EE / 64), 256, 0, stream>>>(
        p_tmp, ao_w + (size_t)l * EE * EE, ao_b + (size_t)l * EE, p_big, M, EE, EE);
    add_ln_kernel<<<M / 8, 256, 0, stream>>>(p_h, p_big, ln1_g + l * EE, ln1_b + l * EE, p_h);
    gemm_bias<1><<<dim3(M / 64, FFD / 64), 256, 0, stream>>>(
        p_h, ff1_w + (size_t)l * FFD * EE, ff1_b + (size_t)l * FFD, p_big, M, FFD, EE);
    gemm_bias<0><<<dim3(M / 64, EE / 64), 256, 0, stream>>>(
        p_big, ff2_w + (size_t)l * EE * FFD, ff2_b + (size_t)l * EE, p_tmp, M, EE, FFD);
    add_ln_kernel<<<M / 8, 256, 0, stream>>>(p_h, p_tmp, ln2_g + l * EE, ln2_b + l * EE, p_h);
  }

  // out = h + x0 ; post MLP -> node_emb (stored in p_h)
  add_kernel<<<(int)(BNE / 256), 256, 0, stream>>>(p_h, p_x0, p_tmp);
  gemm_bias<1><<<dim3(M / 64, EE / 64), 256, 0, stream>>>(p_tmp, post1_w, post1_b, p_big, M, EE, EE);
  gemm_bias<0><<<dim3(M / 64, EE / 64), 256, 0, stream>>>(p_big, post2_w, post2_b, p_h, M, EE, EE);

  // pooled mean, GRU hidden init (tanh), value head
  mean_kernel<<<(BB * EE) / 256, 256, 0, stream>>>(p_h, p_pool);
  gemm_bias<2><<<dim3(BB / 64, EE / 64), 256, 0, stream>>>(p_pool, p2h_w, p2h_b, p_hid, BB, EE, EE);
  gemm_bias<1><<<dim3(BB / 64, 1), 256, 0, stream>>>(p_pool, v1_w, v1_b, p_v1, BB, 64, EE);
  value_kernel<<<1, 256, 0, stream>>>(p_v1, v2_w, v2_b, out + BN + 2 * BB);

  // decode precompute on the matrix cores:
  // PRE = node_emb @ h2q_w  (needs h2q_w transposed to (H,E) "Linear" layout)
  transpose128_kernel<<<(EE * HH) / 256, 256, 0, stream>>>(h2q_w, p_h2qT);
  fill_zero_kernel<<<1, 128, 0, stream>>>(p_zero);
  gemm_bias<0><<<dim3(M / 64, HH / 64), 256, 0, stream>>>(p_h, p_h2qT, p_zero, p_tmp, M, HH, EE);
  // GIPRE = node_emb @ gru_wih^T + gru_bih
  gemm_bias<0><<<dim3(M / 64, 384 / 64), 256, 0, stream>>>(p_h, gru_wih, gru_bih, p_big, M, 384, EE);

  // greedy decode: tours, logp, ent
  decode_kernel<<<BB, 128, 0, stream>>>(p_tmp, p_big, gru_whh, gru_bhh,
                                        p_hid, out, out + BN, out + BN + BB);
}
